// ScaledDotProductAttention_22617297781013
// MI455X (gfx1250) — matrix operand order repaired and checked by reading, not compiled, not run on hardware
//
#include <hip/hip_runtime.h>
#include <hip/hip_bf16.h>
#include <cmath>

typedef __attribute__((ext_vector_type(16))) _Float16 v16h;
typedef __attribute__((ext_vector_type(8)))  _Float16 v8h;
typedef __attribute__((ext_vector_type(4)))  _Float16 v4h;
typedef __attribute__((ext_vector_type(8)))  float    v8f;
typedef __attribute__((ext_vector_type(4)))  float    v4f;

#define S_LEN   2048
#define D_HEAD  64
#define BH      32            // B*H
#define SROW    2052          // padded LDS score-row stride (floats): 2052%64==4 -> conflict-free
#define NWAVE   4
#define WGSIZE  128

// ---------------- prep kernels ----------------

// f32 -> f16 with scale (q gets 1/8 folded in), vectorized x4
__global__ __launch_bounds__(256) void cvt_f16(const float* __restrict__ src,
                                               _Float16* __restrict__ dst,
                                               float scale, int n4) {
  int i = blockIdx.x * blockDim.x + threadIdx.x;
  if (i >= n4) return;
  v4f x = *(const v4f*)(src + (size_t)i * 4);
  v4h y;
  y[0] = (_Float16)(x[0] * scale);
  y[1] = (_Float16)(x[1] * scale);
  y[2] = (_Float16)(x[2] * scale);
  y[3] = (_Float16)(x[3] * scale);
  *(v4h*)(dst + (size_t)i * 4) = y;
}

// vt[bh][d][s] = (f16) v[bh][s][d]  (coalesced writes along s)
__global__ __launch_bounds__(256) void cvt_transpose_v(const float* __restrict__ v,
                                                       _Float16* __restrict__ vt) {
  int s  = blockIdx.x * 256 + threadIdx.x;  // 8 blocks * 256
  int d  = blockIdx.y;                      // 64
  int bh = blockIdx.z;                      // 32
  vt[((size_t)bh * D_HEAD + d) * S_LEN + s] =
      (_Float16)v[((size_t)bh * S_LEN + s) * D_HEAD + d];
}

// ---------------- fused attention ----------------
// grid: (S_LEN/16, BH), block: 128 (4 waves). Each WG: 16 query rows x 2048 keys.
// Dynamic LDS: scores 16*SROW f32 | partials 4*16*64 f32 | reduce 16*8 f32
__global__ __launch_bounds__(WGSIZE) void attn_fused(const _Float16* __restrict__ qh,
                                                     const _Float16* __restrict__ kh,
                                                     const _Float16* __restrict__ vt,
                                                     float* __restrict__ attn,
                                                     float* __restrict__ out) {
  extern __shared__ float lds[];
  float* sS = lds;                       // [16][SROW]
  float* sP = lds + 16 * SROW;           // [4][16][64] partial outputs
  float* sR = sP + NWAVE * 16 * D_HEAD;  // [16][8] reduction scratch

  const int bh   = blockIdx.y;
  const int q0   = blockIdx.x * 16;
  const int tid  = threadIdx.x;
  const int wave = tid >> 5;
  const int lane = tid & 31;
  const int lh   = lane >> 4;   // lane half: 0 or 1
  const int l16  = lane & 15;

  // ---- load Q into WMMA A layout (16x64 = two 16x32 chunks) ----
  // contiguous k order used here for A (B matches): lane(0..15)=row M, K=0..15; lane(16..31)=row M, K=16..31
  const _Float16* qrow =
      qh + ((size_t)(bh * S_LEN + q0 + l16)) * D_HEAD + lh * 16;
  v8h q00 = *(const v8h*)(qrow);
  v8h q01 = *(const v8h*)(qrow + 8);
  v8h q10 = *(const v8h*)(qrow + 32);
  v8h q11 = *(const v8h*)(qrow + 40);
  v16h a0 = __builtin_shufflevector(q00, q01, 0,1,2,3,4,5,6,7,8,9,10,11,12,13,14,15);
  v16h a1 = __builtin_shufflevector(q10, q11, 0,1,2,3,4,5,6,7,8,9,10,11,12,13,14,15);

  // ---- phase 1: scores = (q/8) @ k^T for this wave's 512 keys ----
  const int kt0 = wave * 32;  // 32 key-tiles of 16 keys
  for (int kt = 0; kt < 32; ++kt) {
    const int keyt = (kt0 + kt) * 16;
    // contiguous k order used here for B (matches A): lane(0..15)=col N=key, K(=d)=0..15 ; lane(16..31): K=16..31
    const _Float16* krow =
        kh + ((size_t)(bh * S_LEN + keyt + l16)) * D_HEAD + lh * 16;
    v16h b0 = *(const v16h*)(krow);        // d 0..31 chunk
    v16h b1 = *(const v16h*)(krow + 32);   // d 32..63 chunk
    v8f acc = {};
    acc = __builtin_amdgcn_wmma_f32_16x16x32_f16(false, a0, false, b0,
                                                 (short)0, acc, false, false);
    acc = __builtin_amdgcn_wmma_f32_16x16x32_f16(false, a1, false, b1,
                                                 (short)0, acc, false, false);
    const int col = keyt + l16;
#pragma unroll
    for (int i = 0; i < 8; ++i) {          // C layout: VGPR i -> row i + 8*lh
      sS[(i + lh * 8) * SROW + col] = acc[i];
    }
  }
  __syncthreads();

  // ---- phase 2: softmax over each row (8 threads per row, fp32) ----
  const int r   = tid >> 3;  // 0..15
  const int sub = tid & 7;   // 0..7
  float* rowp = sS + r * SROW;

  float m = -INFINITY;
#pragma unroll 4
  for (int j = 0; j < 64; ++j) {
    v4f x = *(const v4f*)(rowp + (sub + j * 8) * 4);
    m = fmaxf(m, fmaxf(fmaxf(x[0], x[1]), fmaxf(x[2], x[3])));
  }
  sR[r * 8 + sub] = m;
  __syncthreads();
  float rm = sR[r * 8];
#pragma unroll
  for (int i = 1; i < 8; ++i) rm = fmaxf(rm, sR[r * 8 + i]);
  __syncthreads();

  float ssum = 0.f;
#pragma unroll 4
  for (int j = 0; j < 64; ++j) {
    v4f x = *(v4f*)(rowp + (sub + j * 8) * 4);
    v4f e;
    e[0] = __expf(x[0] - rm); e[1] = __expf(x[1] - rm);
    e[2] = __expf(x[2] - rm); e[3] = __expf(x[3] - rm);
    ssum += (e[0] + e[1]) + (e[2] + e[3]);
    *(v4f*)(rowp + (sub + j * 8) * 4) = e;
  }
  sR[r * 8 + sub] = ssum;
  __syncthreads();
  float rs = 0.f;
#pragma unroll
  for (int i = 0; i < 8; ++i) rs += sR[r * 8 + i];
  const float inv = 1.f / rs;

  // normalize in LDS + stream attn to HBM (128B-coalesced per 8-lane group)
  float* gattn = attn + ((size_t)(bh * S_LEN + q0 + r)) * S_LEN;
#pragma unroll 4
  for (int j = 0; j < 64; ++j) {
    const int c = (sub + j * 8) * 4;
    v4f x = *(v4f*)(rowp + c);
    x[0] *= inv; x[1] *= inv; x[2] *= inv; x[3] *= inv;
    *(v4f*)(rowp + c) = x;
    *(v4f*)(gattn + c) = x;
  }
  __syncthreads();

  // ---- phase 3: out = P @ V over this wave's 512 keys ----
  v8f o0 = {}, o1 = {}, o2 = {}, o3 = {};
  for (int kc = 0; kc < 16; ++kc) {
    const int kbase = wave * 512 + kc * 32;  // 32-key chunk
    // A from P (LDS, f32 -> f16), same A layout as above
    const float* prow = sS + l16 * SROW + kbase + lh * 16;
    v4f p0 = *(const v4f*)(prow);
    v4f p1 = *(const v4f*)(prow + 4);
    v4f p2 = *(const v4f*)(prow + 8);
    v4f p3 = *(const v4f*)(prow + 12);
    v16h ap;
#pragma unroll
    for (int i = 0; i < 4; ++i) {
      ap[i]      = (_Float16)p0[i];
      ap[4 + i]  = (_Float16)p1[i];
      ap[8 + i]  = (_Float16)p2[i];
      ap[12 + i] = (_Float16)p3[i];
    }
    // B from V^T: lane holds column d = dt*16 + l16, contiguous 16 keys
    const _Float16* vbase =
        vt + ((size_t)(bh * D_HEAD) + l16) * S_LEN + kbase + lh * 16;
    v16h bv0 = *(const v16h*)(vbase);
    v16h bv1 = *(const v16h*)(vbase + 16 * S_LEN);
    v16h bv2 = *(const v16h*)(vbase + 32 * S_LEN);
    v16h bv3 = *(const v16h*)(vbase + 48 * S_LEN);
    o0 = __builtin_amdgcn_wmma_f32_16x16x32_f16(false, ap, false, bv0, (short)0, o0, false, false);
    o1 = __builtin_amdgcn_wmma_f32_16x16x32_f16(false, ap, false, bv1, (short)0, o1, false, false);
    o2 = __builtin_amdgcn_wmma_f32_16x16x32_f16(false, ap, false, bv2, (short)0, o2, false, false);
    o3 = __builtin_amdgcn_wmma_f32_16x16x32_f16(false, ap, false, bv3, (short)0, o3, false, false);
  }

  // dump per-wave partials, then cross-wave reduce
  float* pw = sP + wave * (16 * D_HEAD);
#pragma unroll
  for (int i = 0; i < 8; ++i) {
    const int row = i + lh * 8;
    pw[row * D_HEAD +  0 + l16] = o0[i];
    pw[row * D_HEAD + 16 + l16] = o1[i];
    pw[row * D_HEAD + 32 + l16] = o2[i];
    pw[row * D_HEAD + 48 + l16] = o3[i];
  }
  __syncthreads();

  for (int e = tid; e < 16 * D_HEAD; e += WGSIZE) {
    const int row = e >> 6, d = e & 63;
    float s = sP[e] + sP[1024 + e] + sP[2048 + e] + sP[3072 + e];
    out[((size_t)(bh * S_LEN + q0 + row)) * D_HEAD + d] = s;
  }
}

// ---------------- launch ----------------
extern "C" void kernel_launch(void* const* d_in, const int* in_sizes, int n_in,
                              void* d_out, int out_size, void* d_ws, size_t ws_size,
                              hipStream_t stream) {
  (void)in_sizes; (void)n_in; (void)out_size; (void)ws_size;
  const float* q = (const float*)d_in[0];
  const float* k = (const float*)d_in[1];
  const float* v = (const float*)d_in[2];

  const size_t n_attn = (size_t)BH * S_LEN * S_LEN;   // 134,217,728
  const size_t n_qkv  = (size_t)BH * S_LEN * D_HEAD;  // 4,194,304
  float* attn = (float*)d_out;
  float* out  = (float*)d_out + n_attn;

  _Float16* qh = (_Float16*)d_ws;
  _Float16* kh = qh + n_qkv;
  _Float16* vt = kh + n_qkv;

  const int n4 = (int)(n_qkv / 4);  // 1,048,576
  cvt_f16<<<dim3(n4 / 256), dim3(256), 0, stream>>>(q, qh, 0.125f, n4);  // fold 1/sqrt(64)
  cvt_f16<<<dim3(n4 / 256), dim3(256), 0, stream>>>(k, kh, 1.0f, n4);
  cvt_transpose_v<<<dim3(S_LEN / 256, D_HEAD, BH), dim3(256), 0, stream>>>(v, vt);

  const size_t lds_bytes = (size_t)(16 * SROW + NWAVE * 16 * D_HEAD + 16 * 8) * sizeof(float);
  attn_fused<<<dim3(S_LEN / 16, BH), dim3(WGSIZE), lds_bytes, stream>>>(qh, kh, vt, attn, out);
}
